// Q_MulIn1Out_Conv1D_34565896798385
// MI455X (gfx1250) — compile-verified, matrix operand order checked by reading, not run on hardware
//
#include <hip/hip_runtime.h>
#include <hip/hip_bf16.h>

typedef __attribute__((ext_vector_type(16))) _Float16 v16h;
typedef __attribute__((ext_vector_type(8)))  _Float16 v8h;
typedef __attribute__((ext_vector_type(8)))  float    v8f;

#define DIMQ 1024
#define NQ 10
#define NW 4092
#define NROWS 65472        // 16*4092
#define ROWS_WG 64
#define N_WGS 1023         // 65472 / 64

// ---------------------------------------------------------------------------
// Kernel 0: Eh = f16(E);  Wh = f16(M*E) where M = [[c,-s],[-s,-c]] (x) I_512
// ---------------------------------------------------------------------------
__global__ void k0_build_EW(const float* __restrict__ E,
                            const float* __restrict__ theta,
                            _Float16* __restrict__ Eh,
                            _Float16* __restrict__ Wh) {
  int idx = blockIdx.x * blockDim.x + threadIdx.x;   // 0 .. 1024*1024-1
  int k = idx >> 10;
  int a = idx & (DIMQ - 1);
  float ct = cosf(theta[0]);
  float st = sinf(theta[0]);
  float e = E[idx];
  float w;
  if (k < 512) w =  ct * e - st * E[(k + 512) * DIMQ + a];
  else         w = -st * E[(k - 512) * DIMQ + a] - ct * e;
  Eh[idx] = (_Float16)e;
  Wh[idx] = (_Float16)w;
}

// ---------------------------------------------------------------------------
// Kernel 1: G = E^T * W  (1024x1024x1024, f16 WMMA, f32 accum, f16 out)
// Computed as D[m=b, n=a] = sum_k W[k,b] * E[k,a]  (G symmetric -> G[a,b])
// LDS-transposed staging panels; row stride 72 halfs = 36 DW: 36*r mod 64
// distinct for r=0..15 and 144B is 16B-aligned -> conflict-free ds b128.
// ---------------------------------------------------------------------------
#define K1S 72
__global__ void k1_gemm_G(const _Float16* __restrict__ Eh,
                          const _Float16* __restrict__ Wh,
                          _Float16* __restrict__ Gh) {
  __shared__ _Float16 Et[64 * K1S];
  __shared__ _Float16 Wt[64 * K1S];
  int t = threadIdx.x;
  int aBase = (blockIdx.x & 15) * 64;
  int bBase = (blockIdx.x >> 4) * 64;
  int wave = t >> 5, lane = t & 31;
  int lrow = lane & 15, lhalf = lane >> 4;
  int at  = wave & 3;           // a-tile 0..3 (WMMA N)
  int bt0 = (wave >> 2) * 2;    // b-tiles bt0, bt0+1 (WMMA M)
  int klo = lhalf * 8, khi = 16 + lhalf * 8;
  int kl = t >> 3;              // 0..31 : k within panel
  int r0 = (t & 7) * 8;         // 0..56 : first of 8 rows this thread stages
  v8f acc0 = {}, acc1 = {};
  for (int kp = 0; kp < DIMQ; kp += 32) {
    v8h ev = *(const v8h*)(Eh + (kp + kl) * DIMQ + aBase + r0);
    v8h wv = *(const v8h*)(Wh + (kp + kl) * DIMQ + bBase + r0);
#pragma unroll
    for (int i = 0; i < 8; ++i) {
      Et[(r0 + i) * K1S + kl] = ev[i];
      Wt[(r0 + i) * K1S + kl] = wv[i];
    }
    __syncthreads();
    union { v16h v; v8h h[2]; } B, A0, A1;
    const _Float16* bp  = Et + (at * 16 + lrow) * K1S;
    const _Float16* ap0 = Wt + ((bt0 + 0) * 16 + lrow) * K1S;
    const _Float16* ap1 = Wt + ((bt0 + 1) * 16 + lrow) * K1S;
    B.h[0]  = *(const v8h*)(bp  + klo);  B.h[1]  = *(const v8h*)(bp  + khi);
    A0.h[0] = *(const v8h*)(ap0 + klo);  A0.h[1] = *(const v8h*)(ap0 + khi);
    A1.h[0] = *(const v8h*)(ap1 + klo);  A1.h[1] = *(const v8h*)(ap1 + khi);
    acc0 = __builtin_amdgcn_wmma_f32_16x16x32_f16(false, A0.v, false, B.v,
                                                  (short)0, acc0, false, false);
    acc1 = __builtin_amdgcn_wmma_f32_16x16x32_f16(false, A1.v, false, B.v,
                                                  (short)0, acc1, false, false);
    __syncthreads();
  }
  // D layout: lane column n = a-tile row (lrow); VGPR v -> m = b = v + 8*lhalf
  v8h o0, o1;
#pragma unroll
  for (int i = 0; i < 8; ++i) { o0[i] = (_Float16)acc0[i]; o1[i] = (_Float16)acc1[i]; }
  _Float16* grow = Gh + (aBase + at * 16 + lrow) * DIMQ + bBase;
  *(v8h*)(grow + (bt0 + 0) * 16 + 8 * lhalf) = o0;
  *(v8h*)(grow + (bt0 + 1) * 16 + 8 * lhalf) = o1;
}

// ---------------------------------------------------------------------------
// Kernel 3: fused quad-form. Per WG: 64 rows.
//   prologue: cos/sin table from x, build 64x1024 psi strip in LDS (f16)
//   main:     D[b,n] = sum_a G[b,a] * psi[n,a] via v_wmma_f32_16x16x32_f16
//   epilogue: z[n] += sum_b D[b,n]*psi[n,b]  (ds b128 reload + ds_add_f32)
// psi row stride 1032 halfs = 2064 B = 516 DW (== 4 mod 64): b128 reads of 16
// different rows cover banks 4r..4r+3 -> conflict-free.
// ---------------------------------------------------------------------------
#define LDA 1032
__global__ void k3_quadform(const float* __restrict__ x,
                            const _Float16* __restrict__ Gh,
                            float* __restrict__ z) {
  extern __shared__ char smem[];
  _Float16* As = (_Float16*)smem;                                   // 64*1032 halfs
  float* csS = (float*)(smem + (size_t)ROWS_WG * LDA * sizeof(_Float16)); // 64*10*2
  float* zS  = csS + ROWS_WG * NQ * 2;                              // 64
  int t = threadIdx.x;
  int rowBase = blockIdx.x * ROWS_WG;

  // 1. per-row cos/sin of 0.5*cols
  if (t < ROWS_WG) {
    int n = rowBase + t;
    int b = n / NW;
    int w = n - b * NW;
    const float* xb = x + (size_t)b * (2 * 4096);
#pragma unroll
    for (int j = 0; j < NQ; ++j) {
      int c = j / 5, kk = j - c * 5;
      float h = 0.5f * xb[c * 4096 + w + kk];
      csS[(t * NQ + j) * 2 + 0] = __cosf(h);
      csS[(t * NQ + j) * 2 + 1] = __sinf(h);
    }
    zS[t] = 0.0f;
  }
  __syncthreads();

  // 2. build psi strip: qubit i maps to bit (9-i) of column index a
  {
    int r = t >> 2;          // row 0..63
    int chunk = t & 3;       // bits 9,8 of a: qubit0 = chunk>>1, qubit1 = chunk&1
    const float* cs = csS + r * NQ * 2;
    float pref = ((chunk & 2) ? cs[1] : cs[0]) * ((chunk & 1) ? cs[3] : cs[2]);
    float cc[8], ss[8];
#pragma unroll
    for (int q = 0; q < 8; ++q) { cc[q] = cs[(q + 2) * 2]; ss[q] = cs[(q + 2) * 2 + 1]; }
    _Float16* dst = As + r * LDA + chunk * 256;
    for (int la = 0; la < 256; ++la) {
      float p = pref;
#pragma unroll
      for (int q = 0; q < 8; ++q) p *= ((la >> (7 - q)) & 1) ? ss[q] : cc[q];
      dst[la] = (_Float16)p;
    }
  }
  __syncthreads();

  // 3. WMMA sweep: wave covers 8 column-tiles of G, 4 row-subtiles of psi
  int wave = t >> 5, lane = t & 31;
  int lrow = lane & 15, lhalf = lane >> 4;
  int klo = lhalf * 8, khi = 16 + lhalf * 8;
  for (int jj = 0; jj < 8; ++jj) {
    int btile = (wave * 8 + jj) * 16;
    const _Float16* Grow = Gh + (size_t)(btile + lrow) * DIMQ;  // A: G row (symmetry)
    v8f acc[4] = {{}, {}, {}, {}};
    for (int k = 0; k < DIMQ; k += 32) {
      union { v16h v; v8h h[2]; } A;
      A.h[0] = *(const v8h*)(Grow + k + klo);
      A.h[1] = *(const v8h*)(Grow + k + khi);
#pragma unroll
      for (int nt = 0; nt < 4; ++nt) {
        union { v16h v; v8h h[2]; } B;
        const _Float16* pr = As + (nt * 16 + lrow) * LDA + k;
        B.h[0] = *(const v8h*)(pr + klo);
        B.h[1] = *(const v8h*)(pr + khi);
        acc[nt] = __builtin_amdgcn_wmma_f32_16x16x32_f16(false, A.v, false, B.v,
                                                         (short)0, acc[nt], false, false);
      }
    }
    // epilogue: lane holds D[b = v + 8*lhalf (within tile), n = lrow]
#pragma unroll
    for (int nt = 0; nt < 4; ++nt) {
      int nloc = nt * 16 + lrow;
      v8h pv = *(const v8h*)(As + nloc * LDA + btile + 8 * lhalf);
      float part = 0.0f;
#pragma unroll
      for (int v = 0; v < 8; ++v) part += acc[nt][v] * (float)pv[v];
      atomicAdd(&zS[nloc], part);   // ds_add_f32
    }
  }
  __syncthreads();
  if (t < ROWS_WG) z[rowBase + t] = zS[t];
}

// ---------------------------------------------------------------------------
extern "C" void kernel_launch(void* const* d_in, const int* in_sizes, int n_in,
                              void* d_out, int out_size, void* d_ws, size_t ws_size,
                              hipStream_t stream) {
  const float* x     = (const float*)d_in[0];   // (16, 2, 4096)
  const float* E     = (const float*)d_in[1];   // (1024, 1024)
  const float* theta = (const float*)d_in[2];   // (10,)
  float* z = (float*)d_out;                     // 65472 floats

  char* ws = (char*)d_ws;
  _Float16* Eh = (_Float16*)ws;                                   // 2 MB
  _Float16* Wh = (_Float16*)(ws + (size_t)DIMQ * DIMQ * 2);       // 2 MB
  _Float16* Gh = (_Float16*)(ws + (size_t)DIMQ * DIMQ * 4);       // 2 MB

  k0_build_EW<<<(DIMQ * DIMQ) / 256, 256, 0, stream>>>(E, theta, Eh, Wh);
  k1_gemm_G<<<256, 256, 0, stream>>>(Eh, Wh, Gh);
  size_t smem = (size_t)ROWS_WG * LDA * sizeof(_Float16)
              + (size_t)ROWS_WG * NQ * 2 * sizeof(float)
              + (size_t)ROWS_WG * sizeof(float);                  // ~134 KB
  k3_quadform<<<N_WGS, 256, smem, stream>>>(x, Gh, z);
}
